// ByteEncoder_51883204936063
// MI455X (gfx1250) — matrix operand order repaired and checked by reading, not compiled, not run on hardware
//
#include <hip/hip_runtime.h>
#include <hip/hip_bf16.h>

typedef __attribute__((ext_vector_type(16))) __bf16 v16bf;
typedef __attribute__((ext_vector_type(8)))  float  v8f;

union AF { v16bf v; uint4 q[2]; };
union BFrag { v16bf v; uint4 q[2]; };

__device__ __forceinline__ unsigned short f2bf(float f) {
    unsigned u = __builtin_bit_cast(unsigned, f);
    unsigned r = u + 0x7FFFu + ((u >> 16) & 1u);
    return (unsigned short)(r >> 16);
}

__device__ __forceinline__ v8f zero8() {
    v8f z;
#pragma unroll
    for (int i = 0; i < 8; ++i) z[i] = 0.0f;
    return z;
}

// ---------------- prep: f32 -> bf16 convert ----------------
__global__ __launch_bounds__(256) void cvt_bf16_kernel(const float* __restrict__ in,
                                                       unsigned short* __restrict__ out, int n) {
    int i = blockIdx.x * 256 + threadIdx.x;
    if (i < n) out[i] = f2bf(in[i]);
}

// conv_w[o][d][k] (o*4096 + d*4 + k) -> Wc[o][k*1024 + d] bf16
__global__ __launch_bounds__(256) void cvt_convw_kernel(const float* __restrict__ cw,
                                                        unsigned short* __restrict__ out) {
    int i = blockIdx.x * 256 + threadIdx.x;      // 0 .. 4M-1
    int o   = i >> 12;
    int rem = i & 4095;
    int k   = rem >> 10;
    int d   = rem & 1023;
    out[i] = f2bf(cw[(o << 12) + (d << 2) + k]);
}

// ---------------- GEMM1: gather-A conv GEMM + bias -> f32 ----------------
// Block tile 128x128 (8 waves as 2x4); wave tile 64x32 (4 row-groups x 2 n-tiles).
// A[m][kk*1024+d] = embed_bf16[x[4m+kk]][d], K = 4096.
__global__ __launch_bounds__(256) void gemm1_conv_kernel(
    const int* __restrict__ x,                 // [32768]
    const unsigned short* __restrict__ E,      // embed bf16 [256][1024]
    const unsigned short* __restrict__ Wc,     // [1024][4096]
    const float* __restrict__ conv_b,
    float* __restrict__ YF)                    // [8192][1024] f32
{
    const int bm   = blockIdx.x >> 3;          // 0..63
    const int bn   = blockIdx.x & 7;           // 0..7
    const int wave = threadIdx.x >> 5;
    const int lane = threadIdx.x & 31;
    const int r    = lane & 15;
    const int hi   = lane >> 4;
    const int wm   = wave >> 2;                // 0..1
    const int wn   = wave & 3;                 // 0..3
    const int m0   = bm * 128 + wm * 64;
    const int n0   = bn * 128 + wn * 32;

    int xi[4][4];
#pragma unroll
    for (int g = 0; g < 4; ++g) {
        int row = m0 + g * 16 + r;
#pragma unroll
        for (int kk = 0; kk < 4; ++kk) xi[g][kk] = x[row * 4 + kk];
    }

    const unsigned short* bcol0 = Wc + (size_t)(n0 + r) * 4096 + hi * 8;
    const unsigned short* bcol1 = Wc + (size_t)(n0 + 16 + r) * 4096 + hi * 8;

    v8f acc[4][2];
#pragma unroll
    for (int g = 0; g < 4; ++g) { acc[g][0] = zero8(); acc[g][1] = zero8(); }

#pragma unroll
    for (int kk = 0; kk < 4; ++kk) {
        const unsigned short* ar0 = E + (size_t)xi[0][kk] * 1024;
        const unsigned short* ar1 = E + (size_t)xi[1][kk] * 1024;
        const unsigned short* ar2 = E + (size_t)xi[2][kk] * 1024;
        const unsigned short* ar3 = E + (size_t)xi[3][kk] * 1024;
        const int kb = kk << 10;
        for (int d0 = 0; d0 < 1024; d0 += 32) {
            AF a[4];
            a[0].q[0] = *(const uint4*)(ar0 + d0 + hi * 8);
            a[0].q[1] = *(const uint4*)(ar0 + d0 + 16 + hi * 8);
            a[1].q[0] = *(const uint4*)(ar1 + d0 + hi * 8);
            a[1].q[1] = *(const uint4*)(ar1 + d0 + 16 + hi * 8);
            a[2].q[0] = *(const uint4*)(ar2 + d0 + hi * 8);
            a[2].q[1] = *(const uint4*)(ar2 + d0 + 16 + hi * 8);
            a[3].q[0] = *(const uint4*)(ar3 + d0 + hi * 8);
            a[3].q[1] = *(const uint4*)(ar3 + d0 + 16 + hi * 8);
            BFrag b0, b1;
            const int k0 = kb + d0;
            b0.q[0] = *(const uint4*)(bcol0 + k0);
            b0.q[1] = *(const uint4*)(bcol0 + k0 + 16);
            b1.q[0] = *(const uint4*)(bcol1 + k0);
            b1.q[1] = *(const uint4*)(bcol1 + k0 + 16);
            __builtin_prefetch(bcol0 + k0 + 64, 0, 3);
            __builtin_prefetch(bcol1 + k0 + 64, 0, 3);
#pragma unroll
            for (int g = 0; g < 4; ++g) {
                acc[g][0] = __builtin_amdgcn_wmma_f32_16x16x32_bf16(
                    false, a[g].v, false, b0.v, (short)0, acc[g][0], false, false);
                acc[g][1] = __builtin_amdgcn_wmma_f32_16x16x32_bf16(
                    false, a[g].v, false, b1.v, (short)0, acc[g][1], false, false);
            }
        }
    }

#pragma unroll
    for (int nt = 0; nt < 2; ++nt) {
        int col = n0 + nt * 16 + r;
        float cbv = conv_b[col];
#pragma unroll
        for (int g = 0; g < 4; ++g) {
#pragma unroll
            for (int i = 0; i < 8; ++i) {
                int row = m0 + g * 16 + i + hi * 8;
                YF[(size_t)row * 1024 + col] = acc[g][nt][i] + cbv;
            }
        }
    }
}

// ---------------- LayerNorm: f32 row -> bf16 row ----------------
__global__ __launch_bounds__(256) void ln_kernel(const float* __restrict__ YF,
                                                 const float* __restrict__ ln_g,
                                                 const float* __restrict__ ln_b,
                                                 unsigned short* __restrict__ Y) {
    __shared__ float ss[8], qq[8];
    const int row  = blockIdx.x;
    const int lane = threadIdx.x & 31;
    const int wave = threadIdx.x >> 5;
    const float4 v = ((const float4*)(YF + (size_t)row * 1024))[threadIdx.x];
    float s = v.x + v.y + v.z + v.w;
    float q = v.x * v.x + v.y * v.y + v.z * v.z + v.w * v.w;
#pragma unroll
    for (int m = 1; m < 32; m <<= 1) { s += __shfl_xor(s, m, 32); q += __shfl_xor(q, m, 32); }
    if (lane == 0) { ss[wave] = s; qq[wave] = q; }
    __syncthreads();
    float S = 0.0f, Q = 0.0f;
#pragma unroll
    for (int w = 0; w < 8; ++w) { S += ss[w]; Q += qq[w]; }
    float mu  = S * (1.0f / 1024.0f);
    float var = Q * (1.0f / 1024.0f) - mu * mu;
    float rs  = rsqrtf(var + 1e-5f);
    const int c = threadIdx.x * 4;
    const float4 g4 = ((const float4*)ln_g)[threadIdx.x];
    const float4 b4 = ((const float4*)ln_b)[threadIdx.x];
    unsigned short* o = Y + (size_t)row * 1024 + c;
    o[0] = f2bf((v.x - mu) * rs * g4.x + b4.x);
    o[1] = f2bf((v.y - mu) * rs * g4.y + b4.y);
    o[2] = f2bf((v.z - mu) * rs * g4.z + b4.z);
    o[3] = f2bf((v.w - mu) * rs * g4.w + b4.w);
}

// ---------------- generic GEMM: bf16 A [M][K] x bf16 W [1024][K] + bias -> f32 ----------------
// Block tile 128x128 (8 waves as 2x4); wave tile 64x32.
__global__ __launch_bounds__(256) void gemm_bias_f32_kernel(
    const unsigned short* __restrict__ A,      // [M][K] bf16
    const unsigned short* __restrict__ W,      // [1024][K] bf16 (row o = B column o)
    const float* __restrict__ bias,            // [1024]
    float* __restrict__ out,                   // [M][1024]
    int K)
{
    const int bm   = blockIdx.x >> 3;
    const int bn   = blockIdx.x & 7;
    const int wave = threadIdx.x >> 5;
    const int lane = threadIdx.x & 31;
    const int r    = lane & 15;
    const int hi   = lane >> 4;
    const int wm   = wave >> 2;
    const int wn   = wave & 3;
    const int m0   = bm * 128 + wm * 64;
    const int n0   = bn * 128 + wn * 32;

    const unsigned short* ar0 = A + (size_t)(m0 + 0 * 16 + r) * K;
    const unsigned short* ar1 = A + (size_t)(m0 + 1 * 16 + r) * K;
    const unsigned short* ar2 = A + (size_t)(m0 + 2 * 16 + r) * K;
    const unsigned short* ar3 = A + (size_t)(m0 + 3 * 16 + r) * K;
    const unsigned short* bcol0 = W + (size_t)(n0 + r) * K + hi * 8;
    const unsigned short* bcol1 = W + (size_t)(n0 + 16 + r) * K + hi * 8;

    v8f acc[4][2];
#pragma unroll
    for (int g = 0; g < 4; ++g) { acc[g][0] = zero8(); acc[g][1] = zero8(); }

    for (int k0 = 0; k0 < K; k0 += 32) {
        AF a[4];
        a[0].q[0] = *(const uint4*)(ar0 + k0 + hi * 8);
        a[0].q[1] = *(const uint4*)(ar0 + k0 + 16 + hi * 8);
        a[1].q[0] = *(const uint4*)(ar1 + k0 + hi * 8);
        a[1].q[1] = *(const uint4*)(ar1 + k0 + 16 + hi * 8);
        a[2].q[0] = *(const uint4*)(ar2 + k0 + hi * 8);
        a[2].q[1] = *(const uint4*)(ar2 + k0 + 16 + hi * 8);
        a[3].q[0] = *(const uint4*)(ar3 + k0 + hi * 8);
        a[3].q[1] = *(const uint4*)(ar3 + k0 + 16 + hi * 8);
        BFrag b0, b1;
        b0.q[0] = *(const uint4*)(bcol0 + k0);
        b0.q[1] = *(const uint4*)(bcol0 + k0 + 16);
        b1.q[0] = *(const uint4*)(bcol1 + k0);
        b1.q[1] = *(const uint4*)(bcol1 + k0 + 16);
        __builtin_prefetch(bcol0 + k0 + 64, 0, 3);
        __builtin_prefetch(bcol1 + k0 + 64, 0, 3);
#pragma unroll
        for (int g = 0; g < 4; ++g) {
            acc[g][0] = __builtin_amdgcn_wmma_f32_16x16x32_bf16(
                false, a[g].v, false, b0.v, (short)0, acc[g][0], false, false);
            acc[g][1] = __builtin_amdgcn_wmma_f32_16x16x32_bf16(
                false, a[g].v, false, b1.v, (short)0, acc[g][1], false, false);
        }
    }

#pragma unroll
    for (int nt = 0; nt < 2; ++nt) {
        int col = n0 + nt * 16 + r;
        float cbv = bias[col];
#pragma unroll
        for (int g = 0; g < 4; ++g) {
#pragma unroll
            for (int i = 0; i < 8; ++i) {
                int row = m0 + g * 16 + i + hi * 8;
                out[(size_t)row * 1024 + col] = acc[g][nt][i] + cbv;
            }
        }
    }
}

// ---------------- LRU scan: h[t] = lam*h[t-1] + (1-lam)*xb[t], per (b,d) ----------------
__global__ __launch_bounds__(256) void ssm_scan_kernel(const float* __restrict__ xb,   // [4][2048][1024]
                                                       const float* __restrict__ log_lambda,
                                                       unsigned short* __restrict__ H) // [4][2048][1024] bf16
{
    int idx = blockIdx.x * 256 + threadIdx.x;   // 0..4095
    int b = idx >> 10;
    int d = idx & 1023;
    float lam   = 1.0f / (1.0f + expf(-expf(log_lambda[d])));
    float oneml = 1.0f - lam;
    const float* p = xb + (size_t)b * 2048 * 1024 + d;
    unsigned short* o = H + (size_t)b * 2048 * 1024 + d;
    float h = 0.0f;
    for (int t = 0; t < 2048; t += 8) {
        float c[8];
#pragma unroll
        for (int j = 0; j < 8; ++j) c[j] = oneml * p[(size_t)(t + j) * 1024];
#pragma unroll
        for (int j = 0; j < 8; ++j) {
            h = fmaf(lam, h, c[j]);
            o[(size_t)(t + j) * 1024] = f2bf(h);
        }
    }
}

extern "C" void kernel_launch(void* const* d_in, const int* in_sizes, int n_in,
                              void* d_out, int out_size, void* d_ws, size_t ws_size,
                              hipStream_t stream) {
    const int*   x        = (const int*)  d_in[0];
    const float* embed    = (const float*)d_in[1];
    const float* conv_w   = (const float*)d_in[2];
    const float* conv_b   = (const float*)d_in[3];
    const float* ln_g     = (const float*)d_in[4];
    const float* ln_b     = (const float*)d_in[5];
    const float* log_lam  = (const float*)d_in[6];
    const float* bW       = (const float*)d_in[7];
    const float* bb       = (const float*)d_in[8];
    const float* cW       = (const float*)d_in[9];
    const float* cb       = (const float*)d_in[10];
    float* out = (float*)d_out;

    char* w = (char*)d_ws;
    unsigned short* EMB = (unsigned short*)(w + 0);          // 512 KB
    unsigned short* WC  = (unsigned short*)(w + 524288);     // 8 MB
    unsigned short* WB  = (unsigned short*)(w + 8912896);    // 2 MB
    unsigned short* WCW = (unsigned short*)(w + 11010048);   // 2 MB
    unsigned short* Y   = (unsigned short*)(w + 13107200);   // 16 MB (bf16 y; later reused as H)
    float*          YF  = (float*)         (w + 29884416);   // 32 MB (f32 y; later reused as XB)
    float*          XB  = (float*)         (w + 29884416);   // reuse of YF (dead after LN)
    unsigned short* H   = (unsigned short*)(w + 13107200);   // reuse of Y (dead after GEMM2)

    // prep: bf16 conversions (+ conv_w transpose)
    cvt_bf16_kernel<<<(256 * 1024) / 256, 256, 0, stream>>>(embed, EMB, 256 * 1024);
    cvt_convw_kernel<<<(1024 * 4096) / 256, 256, 0, stream>>>(conv_w, WC);
    cvt_bf16_kernel<<<(1024 * 1024) / 256, 256, 0, stream>>>(bW, WB, 1024 * 1024);
    cvt_bf16_kernel<<<(1024 * 1024) / 256, 256, 0, stream>>>(cW, WCW, 1024 * 1024);

    // GEMM1 (gather + conv + bias) -> YF f32
    gemm1_conv_kernel<<<512, 256, 0, stream>>>(x, EMB, WC, conv_b, YF);

    // LayerNorm -> Y bf16
    ln_kernel<<<8192, 256, 0, stream>>>(YF, ln_g, ln_b, Y);

    // GEMM2: xb = Y * bW^T + bb  (f32)  [XB overwrites YF region — YF is dead]
    gemm_bias_f32_kernel<<<512, 256, 0, stream>>>(Y, WB, bb, XB, 1024);

    // LRU scan -> H bf16 [overwrites Y region — Y is dead]
    ssm_scan_kernel<<<16, 256, 0, stream>>>(XB, log_lam, H);

    // GEMM3: out = H * cW^T + cb  (f32)
    gemm_bias_f32_kernel<<<512, 256, 0, stream>>>(H, WCW, cb, out, 1024);
}
